// SelfAttention_29377576305294
// MI455X (gfx1250) — matrix operand order repaired and checked by reading, not compiled, not run on hardware
//
#include <hip/hip_runtime.h>
#include <hip/hip_bf16.h>

// ---------------- problem constants ----------------
#define BATCH 4
#define CIN   256
#define HDIM  64
#define WDIM  64
#define COUT  256
#define KW    7
#define PADW  3
#define GRP   8
#define CG    32            // out channels per group
#define CPG   32            // in channels per group
#define HWSZ  (HDIM*WDIM)

// ---------------- tiling ----------------
#define TH 8                // tile height (output rows)
#define TW 16               // tile width  (output cols)
#define TP (TH*TW)          // 128 output positions per tile == blockDim
#define HH   (TH + KW - 1)  // 14 halo rows
#define HWID (TW + KW - 1)  // 22 halo cols
#define HS   (HH*HWID)      // 308 halo positions

typedef __attribute__((ext_vector_type(16))) __bf16 v16bf;
typedef __attribute__((ext_vector_type(8)))  __bf16 v8bf;
typedef __attribute__((ext_vector_type(8)))  float  v8f;

__device__ __forceinline__ int imin(int a, int b) { return a < b ? a : b; }

// One 16x16 output tile of the grouped 1x1-conv GEMM:
//   D[mt..mt+15, n0..n0+15] = W[mt.., 0..31] * X[0..31, cols]
// as a single V_WMMA_F32_16X16X32_BF16 (K = 32 == channels per group).
//
// LDS layouts: W row-major [32][32] bf16; X/D position-major [pos][32] bf16.
// 16-bit operand layouts (ISA 7.12.2) then become contiguous vector accesses:
//   A 16x32: lane needs K = {8hi..8hi+7} and {16+8hi..23+8hi}  -> 2x 16B chunks
//   B 32x16: lane needs K = {8hi..8hi+7} and {16+8hi..23+8hi}                     -> 2x 16B chunks
//   C 16x16 f32: vgpr r -> M = r + 8*hi  -> 8 contiguous outputs -> 1x 16B store
// QMAP remaps GEMM column n (0..127) to the inner-halo position of xs.
template<bool QMAP>
__device__ __forceinline__
void proj_tile(const __bf16* __restrict__ Wlds,   // [32][32] bf16 row-major
               const __bf16* __restrict__ Xlds,   // [HS][32] bf16 position-major
               __bf16* __restrict__ Dlds,         // [ncols][32] bf16 position-major
               int mt, int n0, int nlimit)
{
  const int lane = threadIdx.x & 31;
  const int lo   = lane & 15;
  const int hi   = lane >> 4;
  const int n    = n0 + lo;
  const bool nok = (n < nlimit);

  int s;
  if (QMAP) s = ((n >> 4) + PADW) * HWID + (n & 15) + PADW;  // inner region of halo
  else      s = n;
  const int sl = imin(s, HS - 1);   // clamp loads: no EXEC divergence around WMMA

  // A: two contiguous 8-element chunks of the weight row
  const v8bf alo = *(const v8bf*)(Wlds + (mt + lo) * CPG + 8 * hi);
  const v8bf ahi = *(const v8bf*)(Wlds + (mt + lo) * CPG + 8 * hi + 16);
  const v16bf a  = __builtin_shufflevector(alo, ahi,
                     0,1,2,3,4,5,6,7,8,9,10,11,12,13,14,15);
  // B: two contiguous 8-element chunks of the position row (same K order as A)
  const v8bf blo = *(const v8bf*)(Xlds + sl * CPG + 8 * hi);
  const v8bf bhi = *(const v8bf*)(Xlds + sl * CPG + 8 * hi + 16);
  const v16bf bm = __builtin_shufflevector(blo, bhi,
                     0,1,2,3,4,5,6,7,8,9,10,11,12,13,14,15);

  v8f acc = {};
  acc = __builtin_amdgcn_wmma_f32_16x16x32_bf16(false, a, false, bm,
                                                (short)0, acc, false, false);

  v8bf dv;
#pragma unroll
  for (int r = 0; r < 8; ++r) dv[r] = (__bf16)acc[r];
  if (nok) *(v8bf*)(Dlds + n * CPG + mt + 8 * hi) = dv;   // 16B store
}

__global__ __launch_bounds__(TP, 1)
void sasa_fused_kernel(const float* __restrict__ x,
                       const float* __restrict__ w_q,
                       const float* __restrict__ w_k,
                       const float* __restrict__ w_v,
                       const float* __restrict__ row_emb,
                       const float* __restrict__ col_emb,
                       float* __restrict__ out)
{
  __shared__ __align__(16) __bf16 xs  [HS * CPG];       // 19,712 B : x halo (pos-major)
  __shared__ __align__(16) __bf16 wq_s[CG * CPG];       //  2,048 B
  __shared__ __align__(16) __bf16 wk_s[CG * CPG];
  __shared__ __align__(16) __bf16 wv_s[CG * CPG];
  __shared__ __align__(16) __bf16 qs  [TP * CPG];       //  8,192 B : q (pos-major)
  __shared__ __align__(16) __bf16 kvs [HS * CPG];       // 19,712 B : k, reused for v
  __shared__ __align__(16) __bf16 emb_s[KW * KW * CPG]; //  3,136 B : emb (k-major)
  // total ~56.9 KB of LDS

  const int tid = threadIdx.x;
  const int bx  = blockIdx.x;
  const int tx  = bx & 3;          // W / TW = 4
  const int ty  = (bx >> 2) & 7;   // H / TH = 8
  const int g   = (bx >> 5) & 7;   // GRP
  const int b   = bx >> 8;         // BATCH
  const int x0  = tx * TW;
  const int y0  = ty * TH;

  // ---- stage weights (bf16, row-major) ----
  for (int i = tid; i < CG * CPG; i += TP) {
    wq_s[i] = (__bf16)w_q[g * CG * CPG + i];
    wk_s[i] = (__bf16)w_k[g * CG * CPG + i];
    wv_s[i] = (__bf16)w_v[g * CG * CPG + i];
  }
  // ---- stage relative embeddings, k-major: emb_s[kidx][c] ----
  for (int i = tid; i < KW * KW * CPG; i += TP) {
    int kidx = i >> 5, c = i & 31;
    int ki = kidx / KW, kj = kidx - ki * KW;
    int cc = g * CG + c;
    float e = (cc < COUT / 2) ? row_emb[cc * KW + ki]
                              : col_emb[(cc - COUT / 2) * KW + kj];
    emb_s[kidx * CPG + c] = (__bf16)e;
  }
  // ---- stage x halo (zero padded), position-major xs[s][c] ----
  // global reads stay coalesced (inner index = spatial), LDS write is scattered.
  const float* xg = x + ((size_t)b * CIN + (size_t)g * CPG) * HWSZ;
  for (int i = tid; i < CPG * HS; i += TP) {
    int c  = i / HS, s = i - c * HS;
    int iy = s / HWID, ix = s - iy * HWID;
    int ya = y0 + iy - PADW, xa = x0 + ix - PADW;
    float v = 0.f;
    if ((unsigned)ya < (unsigned)HDIM && (unsigned)xa < (unsigned)WDIM)
      v = xg[(size_t)c * HWSZ + ya * WDIM + xa];
    xs[s * CPG + c] = (__bf16)v;
  }
  __syncthreads();

  const int wv32 = tid >> 5;   // wave id (4 wave32's)

  // ---- Q projection: 8 n-tiles x 2 m-tiles over the inner 128 positions ----
  for (int ch = wv32; ch < 16; ch += 4)
    proj_tile<true>(wq_s, xs, qs, (ch & 1) * 16, (ch >> 1) * 16, TP);
  // ---- K projection: 20 n-tiles x 2 m-tiles over 308 halo positions ----
  for (int ch = wv32; ch < 40; ch += 4)
    proj_tile<false>(wk_s, xs, kvs, (ch & 1) * 16, (ch >> 1) * 16, HS);
  __syncthreads();

  // ---- per-thread attention: one output position each ----
  const int px = tid & 15, py = tid >> 4;

  // q for this position: 64 contiguous bytes
  const v16bf q0 = *(const v16bf*)(qs + tid * CPG);
  const v16bf q1 = *(const v16bf*)(qs + tid * CPG + 16);
  float qv[CPG];
#pragma unroll
  for (int c = 0; c < 16; ++c) { qv[c] = (float)q0[c]; qv[c + 16] = (float)q1[c]; }

  float lg[KW * KW];
#pragma unroll
  for (int ki = 0; ki < KW; ++ki) {
#pragma unroll
    for (int kj = 0; kj < KW; ++kj) {
      const int kidx = ki * KW + kj;
      const int s    = (py + ki) * HWID + (px + kj);
      const v16bf k0 = *(const v16bf*)(kvs + s * CPG);
      const v16bf k1 = *(const v16bf*)(kvs + s * CPG + 16);
      const v16bf e0 = *(const v16bf*)(emb_s + kidx * CPG);      // broadcast
      const v16bf e1 = *(const v16bf*)(emb_s + kidx * CPG + 16); // broadcast
      float acc = 0.f;
#pragma unroll
      for (int c = 0; c < 16; ++c) {
        acc = __builtin_fmaf(qv[c],      (float)k0[c] + (float)e0[c], acc);
        acc = __builtin_fmaf(qv[c + 16], (float)k1[c] + (float)e1[c], acc);
      }
      lg[kidx] = acc;
    }
  }

  // softmax over the 49 window positions (register-local, fp32)
  float mx = lg[0];
#pragma unroll
  for (int k2 = 1; k2 < KW * KW; ++k2) mx = fmaxf(mx, lg[k2]);
  float ssum = 0.f;
#pragma unroll
  for (int k2 = 0; k2 < KW * KW; ++k2) { lg[k2] = __expf(lg[k2] - mx); ssum += lg[k2]; }
  const float rs = 1.f / ssum;
#pragma unroll
  for (int k2 = 0; k2 < KW * KW; ++k2) lg[k2] *= rs;

  // ---- V projection reuses the kvs buffer ----
  __syncthreads();
  for (int ch = wv32; ch < 40; ch += 4)
    proj_tile<false>(wv_s, xs, kvs, (ch & 1) * 16, (ch >> 1) * 16, HS);
  __syncthreads();

  // ---- weighted sum: vector value loads, 32 fp32 accumulators ----
  float oacc[CPG];
#pragma unroll
  for (int c = 0; c < CPG; ++c) oacc[c] = 0.f;

#pragma unroll
  for (int ki = 0; ki < KW; ++ki) {
#pragma unroll
    for (int kj = 0; kj < KW; ++kj) {
      const int s   = (py + ki) * HWID + (px + kj);
      const float w = lg[ki * KW + kj];
      const v16bf v0 = *(const v16bf*)(kvs + s * CPG);
      const v16bf v1 = *(const v16bf*)(kvs + s * CPG + 16);
#pragma unroll
      for (int c = 0; c < 16; ++c) {
        oacc[c]      = __builtin_fmaf((float)v0[c], w, oacc[c]);
        oacc[c + 16] = __builtin_fmaf((float)v1[c], w, oacc[c + 16]);
      }
    }
  }

  float* outg = out + (((size_t)b * COUT + (size_t)g * CG) * HDIM + (y0 + py)) * WDIM
                    + (x0 + px);
#pragma unroll
  for (int c = 0; c < CPG; ++c)
    outg[(size_t)c * HWSZ] = oacc[c];
}

extern "C" void kernel_launch(void* const* d_in, const int* in_sizes, int n_in,
                              void* d_out, int out_size, void* d_ws, size_t ws_size,
                              hipStream_t stream)
{
  (void)in_sizes; (void)n_in; (void)out_size; (void)d_ws; (void)ws_size;
  const float* x   = (const float*)d_in[0];
  const float* wq  = (const float*)d_in[1];
  const float* wk  = (const float*)d_in[2];
  const float* wvp = (const float*)d_in[3];
  const float* re  = (const float*)d_in[4];
  const float* ce  = (const float*)d_in[5];
  float* out = (float*)d_out;

  dim3 grid(BATCH * GRP * (HDIM / TH) * (WDIM / TW));   // 1024 workgroups
  dim3 block(TP);                                       // 128 threads = 4 wave32
  hipLaunchKernelGGL(sasa_fused_kernel, grid, block, 0, stream,
                     x, wq, wk, wvp, re, ce, out);
}
